// MultiHeadAttention_39273180955287
// MI455X (gfx1250) — matrix operand order repaired and checked by reading, not compiled, not run on hardware
//
#include <hip/hip_runtime.h>

typedef __attribute__((ext_vector_type(16))) _Float16 v16h;
typedef __attribute__((ext_vector_type(8)))  _Float16 h8;
typedef __attribute__((ext_vector_type(4)))  _Float16 h4;
typedef __attribute__((ext_vector_type(8)))  float    v8f;

#define DMODEL 1024
#define NHEADS 16
#define SEQ    2048
#define HD     64

// ---------- WMMA fragment loaders (gfx1250 wave32 layouts, ISA 7.12.2) ----------

// A fragment (16x32 f16, MxK) from row-major buffer: row rm+lane%16,
// halves k = g*16 + (lane>=16)*8 + o  ->  two contiguous 8-half (16B) loads.
__device__ __forceinline__ v16h ld_a_frag(const _Float16* __restrict__ base,
                                          int ld, int rm, int kc, int lane) {
  const int r  = rm + (lane & 15);
  const int hi = (lane >> 4) & 1;
  const _Float16* p = base + (size_t)r * ld + kc + hi * 8;
  v16h a;
  *reinterpret_cast<h8*>(&a)       = *reinterpret_cast<const h8*>(p);
  *(reinterpret_cast<h8*>(&a) + 1) = *reinterpret_cast<const h8*>(p + 16);
  return a;
}

// B fragment (32x16 f16, KxN) where logical B[k][n] == X[n][k] (X row-major, ld):
// lane holds n = nb+lane%16, b[g*8+o] = X[n][kc + g*16 + (lane>=16)*8 + o] -> two 8-half loads.
__device__ __forceinline__ v16h ld_bt_frag(const _Float16* __restrict__ base,
                                           int ld, int nb, int kc, int lane) {
  const int n  = nb + (lane & 15);
  const int hi = (lane >> 4) & 1;
  const _Float16* p = base + (size_t)n * ld + kc + hi * 8;
  v16h b;
  *reinterpret_cast<h8*>(&b)       = *reinterpret_cast<const h8*>(p);
  *(reinterpret_cast<h8*>(&b) + 1) = *reinterpret_cast<const h8*>(p + 16);
  return b;
}

// ---------- one-time weight prep: Wt[n][k] = (f16) W[k][n] ----------
__global__ __launch_bounds__(256) void transpose_w_kernel(
    const float* __restrict__ W, _Float16* __restrict__ Wt, int K, int N) {
  __shared__ float tile[32][33];
  const int bx = blockIdx.x * 32;           // n base
  const int by = blockIdx.y * 32;           // k base
  const int tx = threadIdx.x & 31;
  const int ty = threadIdx.x >> 5;          // 0..7
  #pragma unroll
  for (int i = 0; i < 32; i += 8)
    tile[ty + i][tx] = W[(size_t)(by + ty + i) * N + bx + tx];
  __syncthreads();
  #pragma unroll
  for (int i = 0; i < 32; i += 8)
    Wt[(size_t)(bx + ty + i) * K + by + tx] = (_Float16)tile[tx][ty + i];
}

// ---------- projection GEMM: Out(f16)[M,N] = A32[M,K] * W + bias ----------
// W given pre-transposed/converted as Bt(f16)[N][K]. 8 waves, 128x128 tile, BK=64.
__global__ __launch_bounds__(256) void proj_gemm_kernel(
    const float* __restrict__ A32, const _Float16* __restrict__ Bt,
    const float* __restrict__ bias, _Float16* __restrict__ Out,
    int M, int N, int K) {
  constexpr int BM = 128, BN = 128, BK = 64;
  __shared__ _Float16 sA[BM][BK];           // 16 KB

  const int tid  = threadIdx.x;
  const int lane = tid & 31;
  const int wave = tid >> 5;
  const int wm   = (wave >> 2) * 64;
  const int wn   = (wave & 3) * 32;
  const int bm   = blockIdx.y * BM;
  const int bn   = blockIdx.x * BN;

  v8f acc[4][2];
  #pragma unroll
  for (int mt = 0; mt < 4; ++mt)
    #pragma unroll
    for (int nt = 0; nt < 2; ++nt)
      #pragma unroll
      for (int j = 0; j < 8; ++j) acc[mt][nt][j] = 0.0f;

  for (int k0 = 0; k0 < K; k0 += BK) {
    // stage + convert A tile (128x64 f32 -> f16)
    #pragma unroll
    for (int i = 0; i < 8; ++i) {           // 2048 chunks of 4 floats
      int slot = tid + i * 256;
      int r = slot >> 4, g = slot & 15;
      const float* gp = A32 + (size_t)(bm + r) * K + k0 + g * 4;
      float4 v = *reinterpret_cast<const float4*>(gp);
      if (k0 + BK < K) __builtin_prefetch(gp + BK, 0, 3);  // global_prefetch_b8
      h4 hv;
      hv[0] = (_Float16)v.x; hv[1] = (_Float16)v.y;
      hv[2] = (_Float16)v.z; hv[3] = (_Float16)v.w;
      *reinterpret_cast<h4*>(&sA[r][g * 4]) = hv;
    }
    __syncthreads();

    #pragma unroll
    for (int ks = 0; ks < 2; ++ks) {
      v16h af[4];
      #pragma unroll
      for (int mt = 0; mt < 4; ++mt)
        af[mt] = ld_a_frag(&sA[0][0], BK, wm + mt * 16, ks * 32, lane);
      v16h bf[2];
      #pragma unroll
      for (int nt = 0; nt < 2; ++nt)   // W fragments straight from global (L2-resident)
        bf[nt] = ld_bt_frag(Bt, K, bn + wn + nt * 16, k0 + ks * 32, lane);
      #pragma unroll
      for (int mt = 0; mt < 4; ++mt)
        #pragma unroll
        for (int nt = 0; nt < 2; ++nt)
          acc[mt][nt] = __builtin_amdgcn_wmma_f32_16x16x32_f16(
              false, af[mt], false, bf[nt], (short)0, acc[mt][nt], false, false);
    }
    __syncthreads();
  }

  #pragma unroll
  for (int mt = 0; mt < 4; ++mt) {
    #pragma unroll
    for (int nt = 0; nt < 2; ++nt) {
      const int gn    = bn + wn + nt * 16 + (lane & 15);
      const float bb  = bias[gn];
      const int rbase = bm + wm + mt * 16 + ((lane >> 4) << 3);
      #pragma unroll
      for (int j = 0; j < 8; ++j)
        Out[(size_t)(rbase + j) * N + gn] = (_Float16)(acc[mt][nt][j] + bb);
    }
  }
}

// ---------- output GEMM: Out(f32) = A16[M,K] * Wo + bias; zero LDS, zero barriers ----
__global__ __launch_bounds__(256) void out_gemm_kernel(
    const _Float16* __restrict__ A16, const _Float16* __restrict__ Bt,
    const float* __restrict__ bias, float* __restrict__ Out,
    int M, int N, int K) {
  const int tid  = threadIdx.x;
  const int lane = tid & 31;
  const int wave = tid >> 5;
  const int wm   = (wave >> 2) * 64;
  const int wn   = (wave & 3) * 32;
  const int bm   = blockIdx.y * 128;
  const int bn   = blockIdx.x * 128;

  v8f acc[4][2];
  #pragma unroll
  for (int mt = 0; mt < 4; ++mt)
    #pragma unroll
    for (int nt = 0; nt < 2; ++nt)
      #pragma unroll
      for (int j = 0; j < 8; ++j) acc[mt][nt][j] = 0.0f;

  for (int k0 = 0; k0 < K; k0 += 32) {
    v16h af[4];
    #pragma unroll
    for (int mt = 0; mt < 4; ++mt)
      af[mt] = ld_a_frag(A16, K, bm + wm + mt * 16, k0, lane);
    v16h bf[2];
    #pragma unroll
    for (int nt = 0; nt < 2; ++nt)
      bf[nt] = ld_bt_frag(Bt, K, bn + wn + nt * 16, k0, lane);
    #pragma unroll
    for (int mt = 0; mt < 4; ++mt)
      #pragma unroll
      for (int nt = 0; nt < 2; ++nt)
        acc[mt][nt] = __builtin_amdgcn_wmma_f32_16x16x32_f16(
            false, af[mt], false, bf[nt], (short)0, acc[mt][nt], false, false);
  }

  #pragma unroll
  for (int mt = 0; mt < 4; ++mt) {
    #pragma unroll
    for (int nt = 0; nt < 2; ++nt) {
      const int gn    = bn + wn + nt * 16 + (lane & 15);
      const float bb  = bias[gn];
      const int rbase = bm + wm + mt * 16 + ((lane >> 4) << 3);
      #pragma unroll
      for (int j = 0; j < 8; ++j)
        Out[(size_t)(rbase + j) * N + gn] = acc[mt][nt][j] + bb;
    }
  }
}

// ---------- Flash attention (causal), one block = (b, h, 64 q rows), 4 waves ----------
__global__ __launch_bounds__(128) void attn_kernel(
    const _Float16* __restrict__ Yq, const _Float16* __restrict__ Yk,
    const _Float16* __restrict__ Yv, _Float16* __restrict__ Oc) {
  const int qt   = blockIdx.x;
  const int h    = blockIdx.y;
  const int b    = blockIdx.z;
  const int tid  = threadIdx.x;
  const int lane = tid & 31;
  const int wave = tid >> 5;

  __shared__ _Float16 sV[64][64];        // async-DMA landing tile (row-major)
  __shared__ _Float16 sVt[HD][64 + 8];   // V transposed [hd][key]
  __shared__ _Float16 sP[4][16][64];     // per-wave P staging (C->A relayout)

  const size_t basebh = (size_t)b * SEQ * DMODEL + (size_t)h * HD;
  const _Float16* Q  = Yq + basebh;
  const _Float16* Kp = Yk + basebh;
  const _Float16* Vp = Yv + basebh;

  const int q0 = qt * 64 + wave * 16;

  v16h aQ[2];
  #pragma unroll
  for (int s = 0; s < 2; ++s) aQ[s] = ld_a_frag(Q, DMODEL, q0, s * 32, lane);

  v8f accO[4];
  #pragma unroll
  for (int t = 0; t < 4; ++t)
    #pragma unroll
    for (int j = 0; j < 8; ++j) accO[t][j] = 0.0f;
  float mrow[8], lrow[8];
  #pragma unroll
  for (int j = 0; j < 8; ++j) { mrow[j] = -3.0e38f; lrow[j] = 0.0f; }

  const float scale = 0.125f;  // 1/sqrt(64)

  for (int jt = 0; jt <= qt; ++jt) {
    const int kbase = jt * 64;
    __syncthreads();  // previous iteration's sV/sVt/sP readers are done

    // async DMA V tile into LDS (ASYNCcnt path); overlaps with QK^T + softmax below
    #pragma unroll
    for (int i = 0; i < 4; ++i) {
      int slot = tid + i * 128;          // 512 chunks of 8 halves
      int r = slot >> 3, g = slot & 7;
      const _Float16* gp = Vp + (size_t)(kbase + r) * DMODEL + g * 8;
      unsigned ldsoff = (unsigned)(size_t)&sV[r][g * 8];
      asm volatile("global_load_async_to_lds_b128 %0, %1, off"
                   :: "v"(ldsoff), "v"(gp) : "memory");
    }

    // scores S = Q * K^T : B-fragments are contiguous row reads of K (global)
    v8f sc[4];
    #pragma unroll
    for (int ns = 0; ns < 4; ++ns) {
      v8f c = {};
      #pragma unroll
      for (int s = 0; s < 2; ++s) {
        v16h bK = ld_bt_frag(Kp, DMODEL, kbase + ns * 16, s * 32, lane);
        c = __builtin_amdgcn_wmma_f32_16x16x32_f16(false, aQ[s], false, bK,
                                                   (short)0, c, false, false);
      }
      if (jt == qt) {  // diagonal tile: causal mask
        const int key = kbase + ns * 16 + (lane & 15);
        const int qr  = q0 + ((lane >> 4) << 3);
        #pragma unroll
        for (int j = 0; j < 8; ++j)
          c[j] = (key > qr + j) ? -1.0e10f : c[j] * scale;
      } else {
        #pragma unroll
        for (int j = 0; j < 8; ++j) c[j] *= scale;
      }
      sc[ns] = c;
    }

    // row max over 4 n-tiles + 16 lanes of the owning half-wave
    float rmax[8];
    #pragma unroll
    for (int j = 0; j < 8; ++j) {
      float v = fmaxf(fmaxf(sc[0][j], sc[1][j]), fmaxf(sc[2][j], sc[3][j]));
      v = fmaxf(v, __shfl_xor(v, 1, 16));
      v = fmaxf(v, __shfl_xor(v, 2, 16));
      v = fmaxf(v, __shfl_xor(v, 4, 16));
      v = fmaxf(v, __shfl_xor(v, 8, 16));
      rmax[j] = v;
    }

    // online softmax rescale
    #pragma unroll
    for (int j = 0; j < 8; ++j) {
      float mn    = fmaxf(mrow[j], rmax[j]);
      float alpha = __expf(mrow[j] - mn);
      mrow[j] = mn;
      lrow[j] *= alpha;
      #pragma unroll
      for (int t = 0; t < 4; ++t) accO[t][j] *= alpha;
    }
    #pragma unroll
    for (int ns = 0; ns < 4; ++ns)
      #pragma unroll
      for (int j = 0; j < 8; ++j)
        sc[ns][j] = __expf(sc[ns][j] - mrow[j]);
    #pragma unroll
    for (int j = 0; j < 8; ++j) {
      float v = sc[0][j] + sc[1][j] + sc[2][j] + sc[3][j];
      v += __shfl_xor(v, 1, 16);
      v += __shfl_xor(v, 2, 16);
      v += __shfl_xor(v, 4, 16);
      v += __shfl_xor(v, 8, 16);
      lrow[j] += v;
    }

    // V tile has landed by now: wait ASYNCcnt, transpose sV -> sVt
    asm volatile("s_wait_asynccnt 0x0" ::: "memory");
    #pragma unroll
    for (int i = 0; i < 4; ++i) {
      int slot = tid + i * 128;
      int r = slot >> 3, g = slot & 7;
      h8 v = *reinterpret_cast<const h8*>(&sV[r][g * 8]);
      #pragma unroll
      for (int j = 0; j < 8; ++j) sVt[g * 8 + j][r] = v[j];
    }

    // stage P (f16) for C->A fragment relayout
    {
      const int m0 = (lane >> 4) << 3;
      const int n  = lane & 15;
      #pragma unroll
      for (int ns = 0; ns < 4; ++ns)
        #pragma unroll
        for (int j = 0; j < 8; ++j)
          sP[wave][m0 + j][ns * 16 + n] = (_Float16)sc[ns][j];
    }
    __syncthreads();

    // O += P * V
    v16h aP[2];
    #pragma unroll
    for (int s = 0; s < 2; ++s) aP[s] = ld_a_frag(&sP[wave][0][0], 64, 0, s * 32, lane);
    #pragma unroll
    for (int t = 0; t < 4; ++t) {
      #pragma unroll
      for (int s = 0; s < 2; ++s) {
        v16h bV = ld_bt_frag(&sVt[0][0], 64 + 8, t * 16, s * 32, lane);
        accO[t] = __builtin_amdgcn_wmma_f32_16x16x32_f16(false, aP[s], false, bV,
                                                         (short)0, accO[t], false, false);
      }
    }
  }

  // finalize and store f16 to concat buffer [B,S,D]
  _Float16* Op = Oc + basebh;
  const int m0 = (lane >> 4) << 3;
  const int n  = lane & 15;
  #pragma unroll
  for (int j = 0; j < 8; ++j) {
    float inv = 1.0f / lrow[j];
    #pragma unroll
    for (int t = 0; t < 4; ++t)
      Op[(size_t)(q0 + m0 + j) * DMODEL + t * 16 + n] = (_Float16)(accO[t][j] * inv);
  }
}

// ---------- host launch ----------
extern "C" void kernel_launch(void* const* d_in, const int* in_sizes, int n_in,
                              void* d_out, int out_size, void* d_ws, size_t ws_size,
                              hipStream_t stream) {
  const float* q  = (const float*)d_in[0];
  const float* k  = (const float*)d_in[1];
  const float* v  = (const float*)d_in[2];
  const float* Wq = (const float*)d_in[3];
  const float* bq = (const float*)d_in[4];
  const float* Wo = (const float*)d_in[5];
  const float* bo = (const float*)d_in[6];

  const int Bsz = in_sizes[0] / (SEQ * DMODEL);  // 4
  const int M = Bsz * SEQ;                       // 8192
  const int N = DMODEL, K = DMODEL;

  _Float16* Yq  = (_Float16*)d_ws;               // f16 projections (shared Wq)
  _Float16* Yk  = Yq + (size_t)M * DMODEL;
  _Float16* Yv  = Yk + (size_t)M * DMODEL;
  _Float16* Ac  = Yv + (size_t)M * DMODEL;       // attention concat output
  _Float16* Wqt = Ac + (size_t)M * DMODEL;       // Wq^T f16 [N][K]
  _Float16* Wot = Wqt + (size_t)DMODEL * DMODEL; // Wo^T f16 [N][K]

  dim3 tgrid(DMODEL / 32, DMODEL / 32);
  transpose_w_kernel<<<tgrid, 256, 0, stream>>>(Wq, Wqt, DMODEL, DMODEL);
  transpose_w_kernel<<<tgrid, 256, 0, stream>>>(Wo, Wot, DMODEL, DMODEL);

  dim3 ggrid(N / 128, M / 128);                  // (8, 64)
  proj_gemm_kernel<<<ggrid, 256, 0, stream>>>(q, Wqt, bq, Yq, M, N, K);
  proj_gemm_kernel<<<ggrid, 256, 0, stream>>>(k, Wqt, bq, Yk, M, N, K);
  proj_gemm_kernel<<<ggrid, 256, 0, stream>>>(v, Wqt, bq, Yv, M, N, K);

  dim3 agrid(SEQ / 64, NHEADS, Bsz);             // (32, 16, 4)
  attn_kernel<<<agrid, dim3(128), 0, stream>>>(Yq, Yk, Yv, Ac);

  out_gemm_kernel<<<ggrid, 256, 0, stream>>>(Ac, Wot, bo, (float*)d_out, M, N, K);
}
